// LatSim_88450556493845
// MI455X (gfx1250) — matrix operand order repaired and checked by reading, not compiled, not run on hardware
//
#include <hip/hip_runtime.h>

#define N_ 4096
#define M_ 2
#define T_ 2
#define F_ 2048
#define D_ 128
#define C_ 16

typedef unsigned short ushort_t;
typedef __attribute__((ext_vector_type(16))) __bf16      v16bf;
typedef __attribute__((ext_vector_type(8)))  float       v8f;
typedef __attribute__((ext_vector_type(4)))  unsigned int u32x4;
typedef __attribute__((ext_vector_type(4)))  float       f32x4;

union BF16x16 {
    ushort_t u[16];
    u32x4    q[2];
    v16bf    v;
};

// Hardware bf16 convert (single v_cvt on gfx1250) instead of bit-twiddled RNE.
__device__ __forceinline__ ushort_t f2bf(float f) {
    __bf16 b = (__bf16)f;
    return __builtin_bit_cast(ushort_t, b);
}

__device__ __forceinline__ void cvt8(const float* p, ushort_t* h) {
    const f32x4* q4 = reinterpret_cast<const f32x4*>(p);
    f32x4 a = q4[0], b = q4[1];
    h[0] = f2bf(a.x); h[1] = f2bf(a.y); h[2] = f2bf(a.z); h[3] = f2bf(a.w);
    h[4] = f2bf(b.x); h[5] = f2bf(b.y); h[6] = f2bf(b.z); h[7] = f2bf(b.w);
}

// Load 16 bf16 as two 16B chunks (p1 = p0+8 contiguous, p0+16 for A-frag halves)
__device__ __forceinline__ v16bf ld16bf(const ushort_t* p0, const ushort_t* p1) {
    BF16x16 t;
    t.q[0] = *reinterpret_cast<const u32x4*>(p0);
    t.q[1] = *reinterpret_cast<const u32x4*>(p1);
    return t.v;
}

__device__ __forceinline__ v8f wmma_bf16(v16bf a, v16bf b, v8f c) {
    return __builtin_amdgcn_wmma_f32_16x16x32_bf16(false, a, false, b,
                                                   (short)0, c, false, false);
}

// ---------------------------------------------------------------------------
// Kernel 0: pre-convert ys -> bf16 Y^T workspace [T, C, N]
// ---------------------------------------------------------------------------
__global__ __launch_bounds__(256) void latsim_yconv(const float* __restrict__ ys,
                                                    ushort_t* __restrict__ yT)
{
    int base = blockIdx.x * 1024 + threadIdx.x;
#pragma unroll
    for (int i = 0; i < 4; ++i) {
        int idx = base + i * 256;               // idx = (t*C + c)*N + n
        int n  = idx & (N_ - 1);
        int tc = idx >> 12;                     // t*C + c
        int t  = tc >> 4, c = tc & 15;
        yT[idx] = f2bf(ys[((size_t)t * N_ + n) * C_ + c]);
    }
}

// ---------------------------------------------------------------------------
// Kernel 1: z[m,t] = x[:,m,:] @ w[m,t]   ->  bf16 z in workspace [M,T,N,D]
// ---------------------------------------------------------------------------
__global__ __launch_bounds__(256) void latsim_proj(const float* __restrict__ x,
                                                   const float* __restrict__ w,
                                                   ushort_t* __restrict__ z)
{
    __shared__ __align__(16) ushort_t wT[D_ * 128];   // [d][kk], bf16, 32KB

    const int tid  = threadIdx.x;
    const int bx   = blockIdx.x;
    const int mt   = bx >> 5, nt = bx & 31;
    const int m    = mt >> 1, t = mt & 1;
    const int n0   = nt * 128;
    const int wave = tid >> 5, lane = tid & 31;
    const int rlo  = lane & 15, half = lane >> 4;

    const float* xrow  = x + (size_t)(n0 + wave * 16 + rlo) * (M_ * F_) + (size_t)m * F_;
    const float* wbase = w + (size_t)((m * T_ + t) * F_) * D_;

    v8f acc[8];
#pragma unroll
    for (int tb = 0; tb < 8; ++tb) acc[tb] = (v8f){0.f,0.f,0.f,0.f,0.f,0.f,0.f,0.f};

    for (int f0 = 0; f0 < F_; f0 += 128) {
        __syncthreads();
        // stage w chunk transposed: wT[d][kk] = bf16(w[f0+kk][d])
        for (int i = 0; i < 64; ++i) {
            int idx = tid + i * 256;
            int kk = idx >> 7, d = idx & 127;
            wT[d * 128 + kk] = f2bf(wbase[(size_t)(f0 + kk) * D_ + d]);
        }
        __syncthreads();

#pragma unroll
        for (int ks = 0; ks < 4; ++ks) {
            const int k0 = f0 + ks * 32;
            BF16x16 af;                         // A-frag direct from global fp32 x
            cvt8(xrow + k0 + 8 * half,      af.u);
            cvt8(xrow + k0 + 8 * half + 16, af.u + 8);
#pragma unroll
            for (int tb = 0; tb < 8; ++tb) {
                const int d = tb * 16 + rlo;
                const ushort_t* bp = &wT[d * 128 + ks * 32 + 8 * half];
                acc[tb] = wmma_bf16(af.v, ld16bf(bp, bp + 16), acc[tb]);
            }
        }
    }

    ushort_t* zrow = z + (size_t)((m * T_ + t) * N_ + n0) * D_;
#pragma unroll
    for (int tb = 0; tb < 8; ++tb)
#pragma unroll
        for (int v = 0; v < 8; ++v) {
            int row = wave * 16 + v + 8 * half;   // C-layout: row = vgpr + 8*laneHalf
            int col = tb * 16 + rlo;
            zrow[(size_t)row * D_ + col] = f2bf(acc[tb][v]);
        }
}

// ---------------------------------------------------------------------------
// Kernel 2: fused  e = z@z^T -> mask diag -> online softmax -> out = e@y
// Y tile staged via CDNA5 async global->LDS DMA (ASYNCcnt).
// ---------------------------------------------------------------------------
__global__ __launch_bounds__(256) void latsim_attn(const ushort_t* __restrict__ z,
                                                   const ushort_t* __restrict__ yT,
                                                   float* __restrict__ out)
{
    __shared__ __align__(16) ushort_t lds_p[128 * 128]; // P strip staging, 32KB
    __shared__ __align__(16) ushort_t lds_yT[C_ * 128]; // Y^T tile, 4KB

    const int tid  = threadIdx.x;
    const int bx   = blockIdx.x;
    const int mt   = bx >> 5, nt = bx & 31;
    const int m    = mt >> 1, t = mt & 1;
    const int n0   = nt * 128;
    const int wave = tid >> 5, lane = tid & 31;
    const int rlo  = lane & 15, half = lane >> 4;

    const ushort_t* zbase = z + (size_t)(m * T_ + t) * N_ * D_;

    // Preload this wave's 4 q A-fragments (K = d = 0..127), reused for 32 tiles
    const ushort_t* qrow = zbase + (size_t)(n0 + wave * 16 + rlo) * D_;
    v16bf qf[4];
#pragma unroll
    for (int k = 0; k < 4; ++k) {
        const ushort_t* p = qrow + k * 32 + 8 * half;
        qf[k] = ld16bf(p, p + 16);
    }

    v8f O = (v8f){0.f,0.f,0.f,0.f,0.f,0.f,0.f,0.f};
    float mrow[8], lrow[8];
#pragma unroll
    for (int v = 0; v < 8; ++v) { mrow[v] = -3.0e38f; lrow[v] = 0.0f; }

    // Per-thread async-DMA assignment: 16B of Y^T per thread covers the 4KB tile
    const int yc   = tid >> 4;          // 0..15  (c index)
    const int yk8  = (tid & 15) * 8;    // 0..120 (kk start)
    const unsigned lds_dst = (unsigned)(unsigned long long)&lds_yT[yc * 128 + yk8];
    const ushort_t* ysrc_base = yT + ((size_t)t * C_ + yc) * N_ + yk8;

    for (int ct = 0; ct < 32; ++ct) {
        const int c0 = ct * 128;

        __syncthreads();   // previous iteration done reading lds_yT
        // ---- async global->LDS: one b128 per thread (gfx1250 ASYNC path) ----
        {
            const ushort_t* gp = ysrc_base + c0;
            asm volatile("global_load_async_to_lds_b128 %0, %1, off"
                         :: "v"(lds_dst), "v"(gp) : "memory");
            asm volatile("s_wait_asynccnt 0" ::: "memory");
        }
        __syncthreads();

        if (ct + 1 < 32)  // pull next key tile toward the caches
            __builtin_prefetch(zbase + (size_t)(c0 + 128 + (tid >> 1)) * D_, 0, 3);

        // ---- S = q @ k^T  (16x128 per wave, bf16 WMMA, fp32 accum) ----
        v8f S[8];
#pragma unroll
        for (int tb = 0; tb < 8; ++tb) {
            v8f s = (v8f){0.f,0.f,0.f,0.f,0.f,0.f,0.f,0.f};
            const ushort_t* krow = zbase + (size_t)(c0 + tb * 16 + rlo) * D_;
#pragma unroll
            for (int k = 0; k < 4; ++k) {
                const ushort_t* p = krow + k * 32 + 8 * half;
                s = wmma_bf16(qf[k], ld16bf(p, p + 16), s);
            }
            S[tb] = s;
        }

        // ---- diagonal mask ----
#pragma unroll
        for (int v = 0; v < 8; ++v) {
            const int grow = n0 + wave * 16 + v + 8 * half;
#pragma unroll
            for (int tb = 0; tb < 8; ++tb) {
                if (grow == c0 + tb * 16 + rlo) S[tb][v] = -3.0e38f;
            }
        }

        // ---- online softmax: per-row stats via 16-lane xor reductions ----
#pragma unroll
        for (int v = 0; v < 8; ++v) {
            float tmax = S[0][v];
#pragma unroll
            for (int tb = 1; tb < 8; ++tb) tmax = fmaxf(tmax, S[tb][v]);
            tmax = fmaxf(tmax, __shfl_xor(tmax, 1, 32));
            tmax = fmaxf(tmax, __shfl_xor(tmax, 2, 32));
            tmax = fmaxf(tmax, __shfl_xor(tmax, 4, 32));
            tmax = fmaxf(tmax, __shfl_xor(tmax, 8, 32));

            const float mn    = fmaxf(mrow[v], tmax);
            const float scale = __expf(mrow[v] - mn);
            float psum = 0.0f;
            const int row = wave * 16 + v + 8 * half;
#pragma unroll
            for (int tb = 0; tb < 8; ++tb) {
                float p = __expf(S[tb][v] - mn);
                psum += p;
                lds_p[row * 128 + tb * 16 + rlo] = f2bf(p);  // wave-private region
            }
            psum += __shfl_xor(psum, 1, 32);
            psum += __shfl_xor(psum, 2, 32);
            psum += __shfl_xor(psum, 4, 32);
            psum += __shfl_xor(psum, 8, 32);

            lrow[v] = lrow[v] * scale + psum;
            mrow[v] = mn;
            O[v]   *= scale;
        }

        // ---- O += P @ Y  (C=16 -> single WMMA tile column) ----
        const ushort_t* prow = &lds_p[(wave * 16 + rlo) * 128];
#pragma unroll
        for (int k = 0; k < 4; ++k) {
            const ushort_t* pa = prow + k * 32 + 8 * half;
            v16bf a = ld16bf(pa, pa + 16);
            const ushort_t* pb = &lds_yT[rlo * 128 + k * 32 + 8 * half];
            v16bf b = ld16bf(pb, pb + 16);
            O = wmma_bf16(a, b, O);
        }
    }

    // ---- normalize and write out [M,T,N,C] fp32 ----
    float* obase = out + (size_t)((m * T_ + t) * N_ + n0) * C_;
#pragma unroll
    for (int v = 0; v < 8; ++v) {
        int row = wave * 16 + v + 8 * half;
        obase[(size_t)row * C_ + rlo] = O[v] / lrow[v];
    }
}

extern "C" void kernel_launch(void* const* d_in, const int* in_sizes, int n_in,
                              void* d_out, int out_size, void* d_ws, size_t ws_size,
                              hipStream_t stream) {
    const float* x  = (const float*)d_in[0];
    const float* ys = (const float*)d_in[1];
    const float* w  = (const float*)d_in[2];
    float* out      = (float*)d_out;

    ushort_t* z  = (ushort_t*)d_ws;              // bf16 z:  M*T*N*D = 4 MB
    ushort_t* yT = z + (size_t)M_ * T_ * N_ * D_; // bf16 Y^T: T*C*N = 256 KB

    const int blocks = M_ * T_ * (N_ / 128);     // 128
    latsim_yconv<<<(T_ * C_ * N_) / 1024, 256, 0, stream>>>(ys, yT);
    latsim_proj<<<blocks, 256, 0, stream>>>(x, w, z);
    latsim_attn<<<blocks, 256, 0, stream>>>(z, yT, out);
}
